// BiRNNLM_30262339568169
// MI455X (gfx1250) — matrix operand order repaired and checked by reading, not compiled, not run on hardware
//
#include <hip/hip_runtime.h>
#include <hip/hip_bf16.h>

typedef _Float16 f16;
typedef _Float16 v16h __attribute__((ext_vector_type(16)));
typedef _Float16 v8h  __attribute__((ext_vector_type(8)));
typedef float    v8f  __attribute__((ext_vector_type(8)));

#define S_LEN 128
#define BATCH 32
#define EMB   150
#define HID   30
#define VOCAB 32000
#define KP    64   // 2*H=60 padded to 64 for WMMA K-chunks of 32

// ---------------------------------------------------------------------------
// Kernel 1: bidirectional RNN scan. One block per direction, sequential over
// t. Weights (180x30 f32), bias, h-state, h_new, and the gathered embedding
// rows live in LDS (~46 KB). The *pre-update* hidden state at token position
// t is exactly what the reference uses for prediction at t (for both LR and
// RL), so we emit it (as f16) into the padded h_cat matrix (4096 x 64):
//   cols [0,30)  = hLR_used,  cols [30,60) = hRL_used,  cols [60,64) = 0.
// ---------------------------------------------------------------------------
__global__ void __launch_bounds__(256) rnn_scan_kernel(
    const int*   __restrict__ tokens,     // (S,B)
    const float* __restrict__ embedding,  // (V,E)
    const float* __restrict__ W_lr, const float* __restrict__ b_lr,
    const float* __restrict__ W_rl, const float* __restrict__ b_rl,
    const float* __restrict__ h0,         // (1,H)
    f16* __restrict__ hcat)               // (S*B, KP)
{
    const int dir = blockIdx.x;           // 0 = left->right, 1 = right->left
    const int tid = threadIdx.x;

    __shared__ float W_s[(EMB + HID) * HID];
    __shared__ float b_s[HID];
    __shared__ float emb_s[BATCH * EMB];
    __shared__ float h_s[BATCH * HID];
    __shared__ float hn_s[BATCH * HID];

    const float* W  = dir ? W_rl : W_lr;
    const float* bb = dir ? b_rl : b_lr;
    for (int i = tid; i < (EMB + HID) * HID; i += 256) W_s[i] = W[i];
    if (tid < HID) b_s[tid] = bb[tid];
    for (int i = tid; i < BATCH * HID; i += 256) h_s[i] = h0[i % HID];
    __syncthreads();

    const int colBase = dir ? HID : 0;
    for (int t = 0; t < S_LEN; ++t) {
        const int tok = dir ? (S_LEN - 1 - t) : t;

        // emit pre-update state for position `tok`
        for (int i = tid; i < BATCH * HID; i += 256) {
            int b = i / HID, j = i % HID;
            hcat[(size_t)(tok * BATCH + b) * KP + colBase + j] = (f16)h_s[i];
        }
        if (dir == 0 && tid < BATCH * 4) {        // zero K-pad columns 60..63
            int b = tid >> 2, j = tid & 3;
            hcat[(size_t)(tok * BATCH + b) * KP + 2 * HID + j] = (f16)0.0f;
        }
        // gather embedding rows for this timestep
        for (int i = tid; i < BATCH * EMB; i += 256) {
            int b = i / EMB, e = i % EMB;
            int id = tokens[tok * BATCH + b];
            emb_s[i] = embedding[(size_t)id * EMB + e];
        }
        __syncthreads();

        // h_new = tanh([emb, h] @ W + b)
        for (int i = tid; i < BATCH * HID; i += 256) {
            int b = i / HID, j = i % HID;
            float acc = b_s[j];
            const float* eb = &emb_s[b * EMB];
            #pragma unroll 5
            for (int e = 0; e < EMB; ++e) acc += eb[e] * W_s[e * HID + j];
            const float* hb = &h_s[b * HID];
            #pragma unroll
            for (int k = 0; k < HID; ++k) acc += hb[k] * W_s[(EMB + k) * HID + j];
            hn_s[i] = tanhf(acc);
        }
        __syncthreads();
        for (int i = tid; i < BATCH * HID; i += 256) h_s[i] = hn_s[i];
        __syncthreads();
    }
}

// ---------------------------------------------------------------------------
// Kernel 2: W_ho (60 x 32000 f32) -> transposed, K-padded f16 (32000 x 64).
// Transposed layout makes each WMMA B-fragment lane two contiguous 16B loads.
// ---------------------------------------------------------------------------
__global__ void __launch_bounds__(256) convert_who_kernel(
    const float* __restrict__ W_ho, f16* __restrict__ wT)
{
    int n = blockIdx.x * 256 + threadIdx.x;
    if (n >= VOCAB) return;
    #pragma unroll
    for (int k = 0; k < KP; ++k) {
        float v = (k < 2 * HID) ? W_ho[(size_t)k * VOCAB + n] : 0.0f;
        wT[(size_t)n * KP + k] = (f16)v;
    }
}

// ---------------------------------------------------------------------------
// Kernel 3: logits = h_cat(4096x64 f16) @ W_hoT + b_ho, via
// v_wmma_f32_16x16x32_f16 (K=64 -> 2 WMMA per 16x16 tile). 8 waves/block,
// each wave owns one 16x16 tile. Both f16 operands are L2-resident; the
// kernel is bounded by the 524 MB f32 logits write into d_out.
// ---------------------------------------------------------------------------
__global__ void __launch_bounds__(256) logits_wmma_kernel(
    const f16*   __restrict__ hcat,   // (4096, 64)
    const f16*   __restrict__ wT,     // (32000, 64)
    const float* __restrict__ b_ho,   // (32000)
    float* __restrict__ out)          // (4096, 32000)
{
    const int lane    = threadIdx.x & 31;
    const int wave    = threadIdx.x >> 5;
    const int tileN   = blockIdx.x * 8 + wave;
    const int n       = tileN * 16 + (lane & 15);
    const int rowBase = blockIdx.y * 16;
    const int m       = rowBase + (lane & 15);
    const int hiA     = (lane >= 16) ? 8  : 0;  // D: M-offset
    const int hiB     = (lane >= 16) ? 16 : 0;  // B: K-offset

    v8f c = {};
    #pragma unroll
    for (int kc = 0; kc < 2; ++kc) {
        const int k0 = kc * 32;
        // A fragment: 16 contiguous halves at K = k0 + hiB
        const f16* ap = hcat + (size_t)m * KP + k0 + hiB;
        v8h alo = *(const v8h*)(ap);
        v8h ahi = *(const v8h*)(ap + 8);
        // B fragment: 16 contiguous halves at K = k0 + hiB
        const f16* bp = wT + (size_t)n * KP + k0 + hiB;
        v8h blo = *(const v8h*)(bp);
        v8h bhi = *(const v8h*)(bp + 8);
        v16h a, b;
        #pragma unroll
        for (int i = 0; i < 8; ++i) {
            a[i] = alo[i]; a[8 + i] = ahi[i];
            b[i] = blo[i]; b[8 + i] = bhi[i];
        }
        c = __builtin_amdgcn_wmma_f32_16x16x32_f16(
                false, a, false, b, (short)0, c, false, false);
    }

    const float bias = b_ho[n];
    #pragma unroll
    for (int r = 0; r < 8; ++r) {
        int mm = rowBase + r + hiA;       // D: VGPR r -> M = r + (lane>=16)*8
        out[(size_t)mm * VOCAB + n] = c[r] + bias;
    }
}

// ---------------------------------------------------------------------------
// Kernel 4: in-place log_softmax per (s,b) row. One block per row; the 128 KB
// row is staged once in LDS (dynamic, 320 KB/WGP budget) so d_out is read
// exactly once and written exactly once.
// ---------------------------------------------------------------------------
__global__ void __launch_bounds__(256) logsoftmax_kernel(float* __restrict__ out)
{
    extern __shared__ float smem[];       // VOCAB row + 256 reduction slots
    float* row = smem;
    float* red = smem + VOCAB;
    const size_t base = (size_t)blockIdx.x * VOCAB;
    const int tid = threadIdx.x;

    float lmax = -3.4e38f;
    for (int i = tid; i < VOCAB; i += 256) {
        float v = out[base + i];
        row[i] = v;
        lmax = fmaxf(lmax, v);
    }
    red[tid] = lmax; __syncthreads();
    for (int s = 128; s > 0; s >>= 1) {
        if (tid < s) red[tid] = fmaxf(red[tid], red[tid + s]);
        __syncthreads();
    }
    const float mx = red[0];
    __syncthreads();

    float lsum = 0.0f;
    for (int i = tid; i < VOCAB; i += 256) lsum += expf(row[i] - mx);
    red[tid] = lsum; __syncthreads();
    for (int s = 128; s > 0; s >>= 1) {
        if (tid < s) red[tid] += red[tid + s];
        __syncthreads();
    }
    const float lse = logf(red[0]) + mx;
    __syncthreads();

    for (int i = tid; i < VOCAB; i += 256) out[base + i] = row[i] - lse;
}

// ---------------------------------------------------------------------------
extern "C" void kernel_launch(void* const* d_in, const int* in_sizes, int n_in,
                              void* d_out, int out_size, void* d_ws, size_t ws_size,
                              hipStream_t stream) {
    const int*   tokens    = (const int*)  d_in[0];
    const float* embedding = (const float*)d_in[1];
    const float* W_lr      = (const float*)d_in[2];
    const float* b_lr      = (const float*)d_in[3];
    const float* W_rl      = (const float*)d_in[4];
    const float* b_rl      = (const float*)d_in[5];
    const float* W_ho      = (const float*)d_in[6];
    const float* b_ho      = (const float*)d_in[7];
    const float* h0        = (const float*)d_in[8];
    float* out = (float*)d_out;

    char* ws = (char*)d_ws;
    f16* hcat = (f16*)ws;                              // 4096*64*2  = 512 KB
    f16* wT   = (f16*)(ws + 524288);                   // 32000*64*2 = 4 MB

    rnn_scan_kernel<<<2, 256, 0, stream>>>(tokens, embedding,
                                           W_lr, b_lr, W_rl, b_rl, h0, hcat);
    convert_who_kernel<<<(VOCAB + 255) / 256, 256, 0, stream>>>(W_ho, wT);

    dim3 grid(VOCAB / (16 * 8), (S_LEN * BATCH) / 16); // 250 x 256 blocks
    logits_wmma_kernel<<<grid, 256, 0, stream>>>(hcat, wT, b_ho, out);

    logsoftmax_kernel<<<S_LEN * BATCH, 256,
                        (VOCAB + 256) * sizeof(float), stream>>>(out);
}
